// CausalSelfAttention_45200235823551
// MI455X (gfx1250) — matrix operand order repaired and checked by reading, not compiled, not run on hardware
//
#include <hip/hip_runtime.h>
#include <hip/hip_bf16.h>
#include <math.h>

// Problem constants (from reference)
#define Dm   1024
#define Hh   16
#define HD   64
#define Ff   32
#define KRr  3
#define VRr  192
#define Bb   2
#define Ss   2048
#define Mtot (Bb * Ss)   // 4096

typedef __bf16 v16bf __attribute__((ext_vector_type(16)));
typedef __bf16 v8bf  __attribute__((ext_vector_type(8)));
typedef float  v8f   __attribute__((ext_vector_type(8)));

__device__ __forceinline__ v8f vzero8() {
  v8f z;
#pragma unroll
  for (int i = 0; i < 8; ++i) z[i] = 0.f;
  return z;
}

__device__ __forceinline__ v8f wmma_bf16(v16bf a, v16bf b, v8f c) {
  return __builtin_amdgcn_wmma_f32_16x16x32_bf16(false, a, false, b, (short)0, c,
                                                 false, false);
}

// A fragment (16x32 K-chunk) from bf16 row-major storage.
// ISA 7.12.2: lane L<16: M=L, elems 0..7 -> K=0..7, 8..15 -> K=16..23
//             lane L>=16: M=L-16, K=8..15 / 24..31
__device__ __forceinline__ v16bf load_a_bf16(const __bf16* A, int lda, int m0,
                                             int k0, int ln, int half) {
  const __bf16* p = A + (size_t)(m0 + ln) * lda + k0 + 8 * half;
  v8bf x0 = *(const v8bf*)p;
  v8bf x1 = *(const v8bf*)(p + 16);
  v16bf a;
#pragma unroll
  for (int i = 0; i < 8; ++i) { a[i] = x0[i]; a[8 + i] = x1[i]; }
  return a;
}

// B fragment (32x16) from "transposed" bf16 storage: Bp[n*ldk + k].
// ISA layout: lane L: N = L%16, elems 0..7 -> K = 8*(L/16)+i, 8..15 -> K = 16+8*(L/16)+i
__device__ __forceinline__ v16bf load_bT_bf16(const __bf16* Bp, int ldk, int n0,
                                              int k0, int ln, int half) {
  const __bf16* p = Bp + (size_t)(n0 + ln) * ldk + k0 + 8 * half;
  v8bf x0 = *(const v8bf*)p;
  v8bf x1 = *(const v8bf*)(p + 16);
  v16bf b;
#pragma unroll
  for (int i = 0; i < 8; ++i) { b[i] = x0[i]; b[8 + i] = x1[i]; }
  return b;
}

// ---------------------------------------------------------------------------
// x (fp32) -> bf16, vectorized 8 elements/thread.
// ---------------------------------------------------------------------------
__global__ void cvtx_kernel(const float* __restrict__ X,
                            __bf16* __restrict__ Xb) {
  size_t idx = (size_t)blockIdx.x * blockDim.x + threadIdx.x;
  const float4* p = (const float4*)(X + idx * 8);
  float4 a = p[0], b = p[1];
  v8bf o;
  o[0] = (__bf16)a.x; o[1] = (__bf16)a.y; o[2] = (__bf16)a.z; o[3] = (__bf16)a.w;
  o[4] = (__bf16)b.x; o[5] = (__bf16)b.y; o[6] = (__bf16)b.z; o[7] = (__bf16)b.w;
  *(v8bf*)(Xb + idx * 8) = o;
}

// Weight prep: W (KxN, fp32) -> WT (NxK, bf16). One-shot, L2-resident after.
__global__ void wtrans_kernel(const float* __restrict__ W,
                              __bf16* __restrict__ WT, int K, int N) {
  int idx = blockIdx.x * blockDim.x + threadIdx.x;   // K*N threads
  int k = idx % K, n = idx / K;
  WT[(size_t)n * K + k] = (__bf16)W[(size_t)k * N + n];
}

// value_decoder (H,VR,HD) -> (H,HD,VR) bf16.
__global__ void vdtrans_kernel(const float* __restrict__ VD,
                               __bf16* __restrict__ VDT) {
  int idx = blockIdx.x * blockDim.x + threadIdx.x;   // H*HD*VR threads
  int r = idx % VRr;
  int d = (idx / VRr) % HD;
  int h = idx / (VRr * HD);
  VDT[((size_t)h * HD + d) * VRr + r] =
      (__bf16)VD[((size_t)h * VRr + r) * HD + d];
}

// ---------------------------------------------------------------------------
// GEMM: C = A(bf16, MxK row-major) * BT(bf16, NxK row-major, pre-transposed).
// No LDS, no barriers, no converts: B fragments are contiguous 32B loads that
// hit L2 (weights are ~MBs vs 192MB L2). 8 waves; wave tile 32(M) x 64(N):
// 8 wmma per k-step sharing 4 B fragments.
// ---------------------------------------------------------------------------
template <typename TOUT>
__global__ void __launch_bounds__(256)
gemm_kernel(const __bf16* __restrict__ A, const __bf16* __restrict__ BT,
            TOUT* __restrict__ C, int M, int N, int K) {
  const int tid = threadIdx.x, wave = tid >> 5, lane = tid & 31;
  const int ln = lane & 15, half = lane >> 4;
  const int nb = blockIdx.x * 64;
  const int m0 = blockIdx.y * 256 + wave * 32;

  v8f acc[2][4];
#pragma unroll
  for (int t = 0; t < 2; ++t)
#pragma unroll
    for (int nt = 0; nt < 4; ++nt) acc[t][nt] = vzero8();

  for (int k0 = 0; k0 < K; k0 += 32) {
    v16bf a0 = load_a_bf16(A, K, m0, k0, ln, half);
    v16bf a1 = load_a_bf16(A, K, m0 + 16, k0, ln, half);
    if (k0 + 32 < K)
      __builtin_prefetch(A + (size_t)(m0 + ln) * K + k0 + 32, 0, 1);
#pragma unroll
    for (int nt = 0; nt < 4; ++nt) {
      v16bf b = load_bT_bf16(BT, K, nb + nt * 16, k0, ln, half);
      acc[0][nt] = wmma_bf16(a0, b, acc[0][nt]);
      acc[1][nt] = wmma_bf16(a1, b, acc[1][nt]);
    }
  }
#pragma unroll
  for (int t = 0; t < 2; ++t)
#pragma unroll
    for (int nt = 0; nt < 4; ++nt)
#pragma unroll
      for (int r = 0; r < 8; ++r) {
        int m = m0 + t * 16 + r + 8 * half;
        C[(size_t)m * N + nb + nt * 16 + ln] = (TOUT)acc[t][nt][r];
      }
}

// ---------------------------------------------------------------------------
// V decode: Vt[b,h,d,s] = sum_r xv[b,s,r] * value_decoder[h,r,d]
// A = xv bf16 (4096x192), BT = VDT[h] bf16 (64x192). Transposed bf16 epilogue
// so flash's P@V B-fragments are contiguous.
// ---------------------------------------------------------------------------
__global__ void __launch_bounds__(256)
vdecode_kernel(const __bf16* __restrict__ XV, const __bf16* __restrict__ VDT,
               __bf16* __restrict__ Vt) {
  const int tid = threadIdx.x, wave = tid >> 5, lane = tid & 31;
  const int ln = lane & 15, half = lane >> 4;
  const int h = blockIdx.y;
  const int m0 = blockIdx.x * 256 + wave * 32;
  const __bf16* BT = VDT + (size_t)h * HD * VRr;

  v8f acc[2][4];
#pragma unroll
  for (int t = 0; t < 2; ++t)
#pragma unroll
    for (int nt = 0; nt < 4; ++nt) acc[t][nt] = vzero8();

  for (int k0 = 0; k0 < VRr; k0 += 32) {
    v16bf a0 = load_a_bf16(XV, VRr, m0, k0, ln, half);
    v16bf a1 = load_a_bf16(XV, VRr, m0 + 16, k0, ln, half);
#pragma unroll
    for (int nt = 0; nt < 4; ++nt) {
      v16bf b = load_bT_bf16(BT, VRr, nt * 16, k0, ln, half);
      acc[0][nt] = wmma_bf16(a0, b, acc[0][nt]);
      acc[1][nt] = wmma_bf16(a1, b, acc[1][nt]);
    }
  }
#pragma unroll
  for (int t = 0; t < 2; ++t)
#pragma unroll
    for (int nt = 0; nt < 4; ++nt)
#pragma unroll
      for (int r = 0; r < 8; ++r) {
        int m = m0 + t * 16 + r + 8 * half;    // global row in [0,4096)
        int b = m >> 11, s = m & (Ss - 1);
        int d = nt * 16 + ln;
        Vt[(((size_t)b * Hh + h) * HD + d) * Ss + s] = (__bf16)acc[t][nt][r];
      }
}

// ---------------------------------------------------------------------------
// Key decode: k[b,h,s,2f{+1}] = sum_r complex( zk[b,s,f,r,:] * kd[f,h,r,:] )
// ---------------------------------------------------------------------------
__global__ void keydec_kernel(const float* __restrict__ zk,
                              const float* __restrict__ kd,
                              float* __restrict__ ktmp) {
  int idx = blockIdx.x * blockDim.x + threadIdx.x;   // 4096*16*32 threads
  int f = idx & 31, h = (idx >> 5) & 15, m = idx >> 9;
  int b = m >> 11, s = m & (Ss - 1);
  const float* z = zk + (size_t)m * (2 * Ff * KRr) + f * (KRr * 2);
  const float* w = kd + (size_t)f * (Hh * KRr * 2) + h * (KRr * 2);
  float kr = 0.f, ki = 0.f;
#pragma unroll
  for (int r = 0; r < KRr; ++r) {
    float zr = z[2 * r], zi = z[2 * r + 1];
    float br = w[2 * r], bi = w[2 * r + 1];
    kr += zr * br - zi * bi;
    ki += zr * bi + zi * br;
  }
  size_t base = (((size_t)b * Hh + h) * Ss + s) * HD + 2 * f;
  ktmp[base] = kr;
  ktmp[base + 1] = ki;
}

// RoPE on q (layout (b,s,h,d) rows of D) -> bf16 (b,h,s,d), with 1/sqrt(HD).
__global__ void rope_q_kernel(const float* __restrict__ qtmp,
                              __bf16* __restrict__ qb) {
  int idx = blockIdx.x * blockDim.x + threadIdx.x;   // 4096*16*32 threads
  int i = idx & 31, h = (idx >> 5) & 15, m = idx >> 9;
  int b = m >> 11, s = m & (Ss - 1);
  float ang = (float)s * powf(10000.0f, -(float)i * (1.0f / 32.0f));
  float c = cosf(ang), sn = sinf(ang);
  const float* p = qtmp + (size_t)m * Dm + h * HD + 2 * i;
  float t1 = p[0], t2 = p[1];
  size_t base = (((size_t)b * Hh + h) * Ss + s) * HD + 2 * i;
  qb[base]     = (__bf16)((t1 * c - t2 * sn) * 0.125f);
  qb[base + 1] = (__bf16)((t1 * sn + t2 * c) * 0.125f);
}

// RoPE on k (layout already (b,h,s,d)) -> bf16, no scale.
__global__ void rope_k_kernel(const float* __restrict__ ktmp,
                              __bf16* __restrict__ kb) {
  int idx = blockIdx.x * blockDim.x + threadIdx.x;   // B*H*S*32 threads
  int i = idx & 31;
  int row = idx >> 5;                                // (b*H+h)*S + s
  int s = row & (Ss - 1);
  float ang = (float)s * powf(10000.0f, -(float)i * (1.0f / 32.0f));
  float c = cosf(ang), sn = sinf(ang);
  const float* p = ktmp + (size_t)row * HD + 2 * i;
  float t1 = p[0], t2 = p[1];
  kb[(size_t)row * HD + 2 * i]     = (__bf16)(t1 * c - t2 * sn);
  kb[(size_t)row * HD + 2 * i + 1] = (__bf16)(t1 * sn + t2 * c);
}

// ---------------------------------------------------------------------------
// Flash attention. Grid (S/64, B*H), 128 threads (4 waves).
// Wave owns 16 query rows; 64-key tiles with online softmax.
// Q,K bf16 (b,h,s,d); V bf16 transposed (b,h,d,s); Y bf16 (b,s,h*d).
// ---------------------------------------------------------------------------
__global__ void __launch_bounds__(128)
flash_kernel(const __bf16* __restrict__ Q, const __bf16* __restrict__ Kk,
             const __bf16* __restrict__ Vt, __bf16* __restrict__ Y) {
  __shared__ __align__(16) __bf16 Pbuf[4][16 * 64];
  const int tid = threadIdx.x, wave = tid >> 5, lane = tid & 31;
  const int ln = lane & 15, half = lane >> 4;
  const int bh = blockIdx.y, b = bh >> 4, h = bh & 15;
  const int q0 = blockIdx.x * 64 + wave * 16;
  const __bf16* Qp = Q + (size_t)bh * Ss * HD;
  const __bf16* Kp = Kk + (size_t)bh * Ss * HD;
  const __bf16* Vp = Vt + (size_t)bh * HD * Ss;
  __bf16* P = Pbuf[wave];

  v16bf qa[2];
#pragma unroll
  for (int kc = 0; kc < 2; ++kc)
    qa[kc] = load_a_bf16(Qp, HD, q0, kc * 32, ln, half);

  v8f o[4];
#pragma unroll
  for (int nt = 0; nt < 4; ++nt) o[nt] = vzero8();
  float rmax[8], rsum[8];
#pragma unroll
  for (int r = 0; r < 8; ++r) { rmax[r] = -__builtin_inff(); rsum[r] = 0.f; }

  const int ktmax = (q0 + 15) >> 6;
  for (int kt = 0; kt <= ktmax; ++kt) {
    const int kbase = kt * 64;
    // S = Q @ K^T : B fragment = K^T, contiguous per key row of K.
    v8f sc[4];
#pragma unroll
    for (int nt = 0; nt < 4; ++nt) {
      v8f c = vzero8();
#pragma unroll
      for (int kc = 0; kc < 2; ++kc)
        c = wmma_bf16(qa[kc],
                      load_bT_bf16(Kp, HD, kbase + nt * 16, kc * 32, ln, half),
                      c);
      sc[nt] = c;
    }
    // Causal mask (only the last tile can cross the diagonal).
    if (kt == ktmax) {
#pragma unroll
      for (int nt = 0; nt < 4; ++nt)
#pragma unroll
        for (int r = 0; r < 8; ++r) {
          int key = kbase + nt * 16 + ln;
          int row = q0 + r + 8 * half;
          sc[nt][r] = (key > row) ? -__builtin_inff() : sc[nt][r];
        }
    }
    // Online softmax: row stats across the 16-lane half holding each row.
#pragma unroll
    for (int r = 0; r < 8; ++r) {
      float mloc = fmaxf(fmaxf(sc[0][r], sc[1][r]), fmaxf(sc[2][r], sc[3][r]));
#pragma unroll
      for (int msk = 1; msk < 16; msk <<= 1)
        mloc = fmaxf(mloc, __shfl_xor(mloc, msk, 32));
      float mnew = fmaxf(rmax[r], mloc);
      float corr = exp2f((rmax[r] - mnew) * 1.44269504f);
      float ps = 0.f;
#pragma unroll
      for (int nt = 0; nt < 4; ++nt) {
        float p = exp2f((sc[nt][r] - mnew) * 1.44269504f);
        ps += p;
        P[(r + 8 * half) * 64 + nt * 16 + ln] = (__bf16)p;  // C->A relayout
      }
#pragma unroll
      for (int msk = 1; msk < 16; msk <<= 1) ps += __shfl_xor(ps, msk, 32);
      rsum[r] = rsum[r] * corr + ps;
      rmax[r] = mnew;
#pragma unroll
      for (int nt = 0; nt < 4; ++nt) o[nt][r] *= corr;
    }
    // O += P @ V (P re-read from wave-private LDS in A layout; same-wave DS
    // ops are in order, no barrier needed).
#pragma unroll
    for (int kc = 0; kc < 2; ++kc) {
      v16bf pa = load_a_bf16(P, 64, 0, kc * 32, ln, half);
#pragma unroll
      for (int nt = 0; nt < 4; ++nt)
        o[nt] = wmma_bf16(
            pa, load_bT_bf16(Vp, Ss, nt * 16, kbase + kc * 32, ln, half),
            o[nt]);
    }
  }
  // Epilogue: divide by rowsum, write Y in (b, s, h*64+d) layout.
#pragma unroll
  for (int nt = 0; nt < 4; ++nt)
#pragma unroll
    for (int r = 0; r < 8; ++r) {
      int row = q0 + r + 8 * half;
      float val = o[nt][r] / rsum[r];
      Y[((size_t)b * Ss + row) * Dm + h * HD + nt * 16 + ln] = (__bf16)val;
    }
}

// ---------------------------------------------------------------------------
extern "C" void kernel_launch(void* const* d_in, const int* in_sizes, int n_in,
                              void* d_out, int out_size, void* d_ws,
                              size_t ws_size, hipStream_t stream) {
  (void)in_sizes; (void)n_in; (void)out_size; (void)ws_size;
  const float* x      = (const float*)d_in[0];
  const float* Wq     = (const float*)d_in[1];
  const float* Wk     = (const float*)d_in[2];
  const float* Wv     = (const float*)d_in[3];
  const float* keydec = (const float*)d_in[4];
  const float* valdec = (const float*)d_in[5];
  const float* Wproj  = (const float*)d_in[6];
  float* out = (float*)d_out;

  const size_t MB = 1024 * 1024;
  char* ws = (char*)d_ws;
  __bf16* xb   = (__bf16*)(ws);              // 8 MB
  float*  qtmp = (float*)(ws + 8 * MB);      // 16 MB (reused as ktmp)
  float*  zk   = (float*)(ws + 24 * MB);     // 3 MB
  __bf16* xvb  = (__bf16*)(ws + 27 * MB);    // 1.5 MB
  __bf16* qb   = (__bf16*)(ws + 29 * MB);    // 8 MB
  __bf16* kb   = (__bf16*)(ws + 37 * MB);    // 8 MB
  __bf16* vtb  = (__bf16*)(ws + 45 * MB);    // 8 MB
  __bf16* yb   = (__bf16*)(ws + 53 * MB);    // 8 MB
  __bf16* wqT  = (__bf16*)(ws + 61 * MB);    // 2 MB   (1024x1024)
  __bf16* wkT  = (__bf16*)(ws + 63 * MB);    // 0.4 MB (192x1024)
  __bf16* wvT  = (__bf16*)(ws + 64 * MB);    // 0.4 MB (192x1024)
  __bf16* wpT  = (__bf16*)(ws + 65 * MB);    // 2 MB   (1024x1024)
  __bf16* vdT  = (__bf16*)(ws + 67 * MB);    // 0.4 MB (16x64x192)

  // 0) One-shot prep: x -> bf16; all weights -> bf16 transposed (NxK).
  cvtx_kernel<<<(Mtot * Dm / 8) / 256, 256, 0, stream>>>(x, xb);
  wtrans_kernel<<<(Dm * Dm) / 256, 256, 0, stream>>>(Wq, wqT, Dm, Dm);
  wtrans_kernel<<<(Dm * 2 * Ff * KRr) / 256, 256, 0, stream>>>(
      Wk, wkT, Dm, 2 * Ff * KRr);
  wtrans_kernel<<<(Dm * VRr) / 256, 256, 0, stream>>>(Wv, wvT, Dm, VRr);
  wtrans_kernel<<<(Dm * Dm) / 256, 256, 0, stream>>>(Wproj, wpT, Dm, Dm);
  vdtrans_kernel<<<(Hh * HD * VRr) / 256, 256, 0, stream>>>(valdec, vdT);

  // 1) q path: qtmp = x @ Wq ; RoPE+scale -> qb (b,h,s,d)
  gemm_kernel<float><<<dim3(Dm / 64, Mtot / 256), 256, 0, stream>>>(
      xb, wqT, qtmp, Mtot, Dm, Dm);
  rope_q_kernel<<<(Mtot * Hh * 32) / 256, 256, 0, stream>>>(qtmp, qb);

  // 2) k path: zk = x @ Wk ; complex decode -> ktmp (reuse qtmp) ; RoPE -> kb
  gemm_kernel<float><<<dim3((2 * Ff * KRr) / 64, Mtot / 256), 256, 0, stream>>>(
      xb, wkT, zk, Mtot, 2 * Ff * KRr, Dm);
  float* ktmp = qtmp;
  keydec_kernel<<<(Mtot * Hh * Ff) / 256, 256, 0, stream>>>(zk, keydec, ktmp);
  rope_k_kernel<<<(Bb * Hh * Ss * 32) / 256, 256, 0, stream>>>(ktmp, kb);

  // 3) v path: xvb = x @ Wv (bf16 out) ; per-head decode -> vtb (b,h,d,s)
  gemm_kernel<__bf16><<<dim3(VRr / 64, Mtot / 256), 256, 0, stream>>>(
      xb, wvT, xvb, Mtot, VRr, Dm);
  vdecode_kernel<<<dim3(Mtot / 256, Hh), 256, 0, stream>>>(xvb, vdT, vtb);

  // 4) flash attention -> yb (b,s,D)
  flash_kernel<<<dim3(Ss / 64, Bb * Hh), 128, 0, stream>>>(qb, kb, vtb, yb);

  // 5) out = yb @ Wproj
  gemm_kernel<float><<<dim3(Dm / 64, Mtot / 256), 256, 0, stream>>>(
      yb, wpT, out, Mtot, Dm, Dm);
}
